// VectorQuantizer_86028194939204
// MI455X (gfx1250) — matrix operand order repaired and checked by reading, not compiled, not run on hardware
//
#include <hip/hip_runtime.h>
#include <cmath>
#include <cstdint>

#define NUM_EMB   8192
#define EMB_DIM   256
#define N_ROWS    32768                 // 8 * 4096
#define CHUNK     32                    // codebook rows per iteration
#define NCHUNK    (NUM_EMB / CHUNK)     // 256
#define WAVES     4
#define ROWS_WG   (WAVES * 16)          // 64 rows per workgroup
#define NWG       (N_ROWS / ROWS_WG)    // 512 workgroups

// workspace layout
//   floats:  [0..8191] ||e||^2 | [8192..8447] colsum(E) | [8448..8959] loss partials
//   bytes :  36864 + 4MB bf16 E row-major | +4MB bf16 E transposed
#define WS_EK2    0
#define WS_ESUM   (WS_EK2 + NUM_EMB)
#define WS_LOSS   (WS_ESUM + EMB_DIM)
#define WS_EBF_B  36864u
#define WS_EBFT_B (WS_EBF_B + 4u * 1024u * 1024u)
#define WS_REQ_B  (WS_EBFT_B + 4u * 1024u * 1024u)

typedef __attribute__((ext_vector_type(16))) __bf16 v16bf;
typedef __attribute__((ext_vector_type(8)))  __bf16 v8bf;
typedef __attribute__((ext_vector_type(8)))  float  v8f;

__device__ __forceinline__ v8f wmma_bf16(v16bf a, v16bf b, v8f c) {
    return __builtin_amdgcn_wmma_f32_16x16x32_bf16(false, a, false, b, (short)0, c,
                                                   false, false);
}

// ---- CDNA5 async global->LDS copy (ASYNCcnt), GVS mode: mem = saddr + voffset ----
__device__ __forceinline__ void async_ld_b128(uint32_t lds_off, uint32_t voff,
                                              const void* sbase) {
    asm volatile("global_load_async_to_lds_b128 %0, %1, %2"
                 :: "v"(lds_off), "v"(voff), "s"(sbase) : "memory");
}
__device__ __forceinline__ void wait_async_le16() {
    asm volatile("s_wait_asynccnt 0x10" ::: "memory");
}
__device__ __forceinline__ void wait_async_le0() {
    asm volatile("s_wait_asynccnt 0x0" ::: "memory");
}

// ---------------- prep: per-row ||e||^2 (fp32) ----------------
__global__ void vq_prep_rows(const float* __restrict__ E, float* __restrict__ ws) {
    const int row = blockIdx.x * blockDim.x + threadIdx.x;   // 8192 threads
    const float4* p = (const float4*)(E + (size_t)row * EMB_DIM);
    float s = 0.f;
    #pragma unroll 4
    for (int i = 0; i < EMB_DIM / 4; ++i) {
        const float4 v = p[i];
        s += v.x * v.x + v.y * v.y + v.z * v.z + v.w * v.w;
    }
    ws[WS_EK2 + row] = s;
}

// ---------------- prep: per-column sums of E (fp32) ----------------
__global__ void vq_prep_cols(const float* __restrict__ E, float* __restrict__ ws) {
    __shared__ float part[4][EMB_DIM];
    const int d = threadIdx.x & 255;
    const int g = threadIdx.x >> 8;     // 0..3
    float s = 0.f;
    for (int n = g; n < NUM_EMB; n += 4) s += E[(size_t)n * EMB_DIM + d];
    part[g][d] = s;
    __syncthreads();
    if (g == 0) ws[WS_ESUM + d] = part[0][d] + part[1][d] + part[2][d] + part[3][d];
}

// ---------------- prep: bf16 copies of E (row-major + transposed) ----------------
__global__ void vq_prep_bf16(const float* __restrict__ E, __bf16* __restrict__ Ebf,
                             __bf16* __restrict__ EbfT) {
    const int base = (blockIdx.x * 256 + threadIdx.x) * 8;   // 1024 blocks x 256 thr
    const int n = base >> 8;
    const int d = base & 255;
    const float4 a = *(const float4*)(E + base);
    const float4 b = *(const float4*)(E + base + 4);
    const __bf16 v[8] = {(__bf16)a.x, (__bf16)a.y, (__bf16)a.z, (__bf16)a.w,
                         (__bf16)b.x, (__bf16)b.y, (__bf16)b.z, (__bf16)b.w};
    #pragma unroll
    for (int i = 0; i < 8; ++i) Ebf[base + i] = v[i];
    #pragma unroll
    for (int i = 0; i < 8; ++i) EbfT[(size_t)(d + i) * NUM_EMB + n] = v[i];
}

// ---------------- shared per-chunk compute (GEMM1 -> expm1 -> GEMM2) ----------------
__device__ __forceinline__ void vq_compute_chunk(
    const __bf16 (* __restrict__ Eld)[EMB_DIM],   // [32][256] chunk, row-major
    const __bf16 (* __restrict__ Etl)[CHUNK],     // [256][32] chunk, transposed
    __bf16 (* __restrict__ Pld)[CHUNK],           // [16][32] this wave's P buffer
    const v16bf* __restrict__ Xa, v8f* __restrict__ O, float* __restrict__ lpart,
    const float* __restrict__ ws, int ch, int l16, int h)
{
    const v8f vzero = {};
    v8f S0 = vzero, S1 = vzero;
    #pragma unroll
    for (int c = 0; c < 8; ++c) {
        const int dbase = 32 * c + h * 16;
        const v16bf B0 = *(const v16bf*)&Eld[l16][dbase];
        const v16bf B1 = *(const v16bf*)&Eld[l16 + 16][dbase];
        S0 = wmma_bf16(Xa[c], B0, S0);
        S1 = wmma_bf16(Xa[c], B1, S1);
    }

    const int nbase = ch * CHUNK;
    const float ek0 = ws[WS_EK2 + nbase + l16];
    const float ek1 = ws[WS_EK2 + nbase + 16 + l16];
    float P0[8], P1[8];
    #pragma unroll
    for (int r = 0; r < 8; ++r) {
        P0[r] = expm1f(2.f * S0[r] - ek0);
        P1[r] = expm1f(2.f * S1[r] - ek1);
        lpart[r] += P0[r] + P1[r];
    }

    // re-layout P' (C/D layout -> A layout) through same-wave LDS
    #pragma unroll
    for (int r = 0; r < 8; ++r) {
        const int M = r + 8 * h;
        Pld[M][l16]      = (__bf16)P0[r];
        Pld[M][16 + l16] = (__bf16)P1[r];
    }
    v16bf Pa;
    {
        const v8bf lo = *(const v8bf*)&Pld[l16][h * 16];
        const v8bf hi = *(const v8bf*)&Pld[l16][h * 16 + 8];
        #pragma unroll
        for (int i = 0; i < 8; ++i) { Pa[i] = lo[i]; Pa[i + 8] = hi[i]; }
    }

    #pragma unroll
    for (int t = 0; t < 16; ++t) {
        const v16bf Bt = *(const v16bf*)&Etl[16 * t + l16][h * 16];
        O[t] = wmma_bf16(Pa, Bt, O[t]);
    }
}

// ---------------- shared epilogue: normalize, store q, reduce loss ----------------
__device__ __forceinline__ void vq_epilogue(
    const float* __restrict__ X, float* __restrict__ ws, float* __restrict__ Out,
    const v8f* __restrict__ O, const float* __restrict__ lpart,
    float* __restrict__ lossW, int row0, int l16, int h, int lane, int wave)
{
    float rinv[8];
    #pragma unroll
    for (int r = 0; r < 8; ++r) {
        float v = lpart[r];
        v += __shfl_xor(v, 1, 32);
        v += __shfl_xor(v, 2, 32);
        v += __shfl_xor(v, 4, 32);
        v += __shfl_xor(v, 8, 32);
        rinv[r] = 1.f / ((float)NUM_EMB + v);
    }

    double lacc = 0.0;
    #pragma unroll
    for (int t = 0; t < 16; ++t) {
        const int d = 16 * t + l16;
        const float esum = ws[WS_ESUM + d];
        #pragma unroll
        for (int r = 0; r < 8; ++r) {
            const int row = row0 + r + 8 * h;
            const float q  = (O[t][r] + esum) * rinv[r];
            const float x  = X[(size_t)row * EMB_DIM + d];
            const float df = q - x;
            lacc += (double)(df * df);
            Out[(size_t)row * EMB_DIM + d] = q;   // straight-through value == q
        }
    }
    lacc += __shfl_xor(lacc, 1, 32);
    lacc += __shfl_xor(lacc, 2, 32);
    lacc += __shfl_xor(lacc, 4, 32);
    lacc += __shfl_xor(lacc, 8, 32);
    lacc += __shfl_xor(lacc, 16, 32);
    if (lane == 0) lossW[wave] = (float)lacc;
    __syncthreads();
    if (threadIdx.x == 0)
        ws[WS_LOSS + blockIdx.x] = lossW[0] + lossW[1] + lossW[2] + lossW[3];
}

// ---------------- load X A-fragments (16x256 bf16, contiguous k = 16h + s) ----------------
__device__ __forceinline__ void vq_load_xa(const float* __restrict__ X, int row0,
                                           int l16, int h, v16bf* __restrict__ Xa) {
    const float* xrow = X + (size_t)(row0 + l16) * EMB_DIM;
    #pragma unroll
    for (int c = 0; c < 8; ++c) {
        const int d0 = 32 * c + (h ? 16 : 0);
        v16bf a;
        #pragma unroll
        for (int i = 0; i < 8; ++i) a[i]     = (__bf16)xrow[d0 + i];
        #pragma unroll
        for (int i = 0; i < 8; ++i) a[i + 8] = (__bf16)xrow[d0 + 8 + i];
        Xa[c] = a;
    }
}

// ================= main kernel, async double-buffered staging =================
__launch_bounds__(128)
__global__ void vq_main_async(const float* __restrict__ X,
                              const __bf16* __restrict__ Ebf,
                              const __bf16* __restrict__ EbfT,
                              float* __restrict__ ws, float* __restrict__ Out)
{
    __shared__ __align__(32) __bf16 Elds[2][CHUNK][EMB_DIM];    // 2 x 16 KB
    __shared__ __align__(32) __bf16 Etlds[2][EMB_DIM][CHUNK];   // 2 x 16 KB
    __shared__ __align__(32) __bf16 Plds[WAVES][16][CHUNK];     // 4 KB
    __shared__ float lossW[WAVES];

    const int tid  = threadIdx.x;
    const int wave = tid >> 5;
    const int lane = tid & 31;
    const int l16  = lane & 15;
    const int h    = lane >> 4;
    const int row0 = blockIdx.x * ROWS_WG + wave * 16;

    const uint32_t eBase  = (uint32_t)(uintptr_t)&Elds[0][0][0];
    const uint32_t etBase = (uint32_t)(uintptr_t)&Etlds[0][0][0];

    v16bf Xa[8];
    vq_load_xa(X, row0, l16, h, Xa);

    const v8f vzero = {};
    v8f O[16];
    #pragma unroll
    for (int t = 0; t < 16; ++t) O[t] = vzero;
    float lpart[8] = {0.f, 0.f, 0.f, 0.f, 0.f, 0.f, 0.f, 0.f};

    // stage chunk `ch` into buffer `buf`: 8 + 8 async b128 copies per thread
    auto stage = [&](int ch, int buf) {
        const char* gR = (const char*)(Ebf  + (size_t)ch * CHUNK * EMB_DIM);
        const char* gT = (const char*)(EbfT + (size_t)ch * CHUNK);
        #pragma unroll
        for (int j = 0; j < 8; ++j) {              // row-major: contiguous 16 KB
            const uint32_t off = (uint32_t)(tid * 16 + j * 2048);
            async_ld_b128(eBase + (uint32_t)buf * 16384u + off, off, gR);
        }
        #pragma unroll
        for (int j = 0; j < 8; ++j) {              // transposed: 256 rows x 64 B
            const int k = tid + 128 * j;
            const int dd = k >> 2, p = k & 3;
            async_ld_b128(etBase + (uint32_t)buf * 16384u + (uint32_t)(dd * 64 + p * 16),
                          (uint32_t)(dd * 16384 + p * 16), gT);
        }
    };

    stage(0, 0);
    for (int ch = 0; ch < NCHUNK; ++ch) {
        const int cur = ch & 1;
        if (ch + 1 < NCHUNK) { stage(ch + 1, cur ^ 1); wait_async_le16(); }
        else                 { wait_async_le0(); }
        __syncthreads();                         // everyone's chunk `ch` data landed
        vq_compute_chunk(Elds[cur], Etlds[cur], Plds[wave], Xa, O, lpart, ws, ch, l16, h);
        __syncthreads();                         // buffer `cur` free for chunk ch+2
    }

    vq_epilogue(X, ws, Out, O, lpart, lossW, row0, l16, h, lane, wave);
}

// ================= fallback main kernel, in-register conversion staging =================
__launch_bounds__(128)
__global__ void vq_main_conv(const float* __restrict__ X, const float* __restrict__ E,
                             float* __restrict__ ws, float* __restrict__ Out)
{
    __shared__ __align__(32) __bf16 Elds[CHUNK][EMB_DIM];
    __shared__ __align__(32) __bf16 Etlds[EMB_DIM][CHUNK];
    __shared__ __align__(32) __bf16 Plds[WAVES][16][CHUNK];
    __shared__ float lossW[WAVES];

    const int tid  = threadIdx.x;
    const int wave = tid >> 5;
    const int lane = tid & 31;
    const int l16  = lane & 15;
    const int h    = lane >> 4;
    const int row0 = blockIdx.x * ROWS_WG + wave * 16;

    v16bf Xa[8];
    vq_load_xa(X, row0, l16, h, Xa);

    const v8f vzero = {};
    v8f O[16];
    #pragma unroll
    for (int t = 0; t < 16; ++t) O[t] = vzero;
    float lpart[8] = {0.f, 0.f, 0.f, 0.f, 0.f, 0.f, 0.f, 0.f};

    for (int ch = 0; ch < NCHUNK; ++ch) {
        __syncthreads();
        const float* Ec = E + (size_t)ch * CHUNK * EMB_DIM;
        #pragma unroll
        for (int jj = 0; jj < 16; ++jj) {
            const int base = (tid + 128 * jj) * 4;
            const int n = base >> 8;
            const int d = base & 255;
            const float4 v = *(const float4*)(Ec + base);
            const __bf16 b0 = (__bf16)v.x, b1 = (__bf16)v.y,
                         b2 = (__bf16)v.z, b3 = (__bf16)v.w;
            Elds[n][d] = b0; Elds[n][d + 1] = b1; Elds[n][d + 2] = b2; Elds[n][d + 3] = b3;
            Etlds[d][n] = b0; Etlds[d + 1][n] = b1; Etlds[d + 2][n] = b2; Etlds[d + 3][n] = b3;
        }
        if (ch + 1 < NCHUNK) {
            const float* En = Ec + CHUNK * EMB_DIM;
            __builtin_prefetch(En + tid * 64, 0, 0);
            __builtin_prefetch(En + tid * 64 + 32, 0, 0);
        }
        __syncthreads();
        vq_compute_chunk(Elds, Etlds, Plds[wave], Xa, O, lpart, ws, ch, l16, h);
    }

    vq_epilogue(X, ws, Out, O, lpart, lossW, row0, l16, h, lane, wave);
}

// ---------------- finalize: deterministic loss reduction ----------------
__global__ void vq_finalize(const float* __restrict__ ws, const float* __restrict__ cc,
                            float* __restrict__ Out) {
    double s = 0.0;
    for (int i = 0; i < NWG; ++i) s += (double)ws[WS_LOSS + i];
    const double numel = (double)N_ROWS * (double)EMB_DIM;
    Out[(size_t)N_ROWS * EMB_DIM] = (float)((1.0 + (double)cc[0]) * s / numel);
}

extern "C" void kernel_launch(void* const* d_in, const int* in_sizes, int n_in,
                              void* d_out, int out_size, void* d_ws, size_t ws_size,
                              hipStream_t stream) {
    (void)in_sizes; (void)n_in; (void)out_size;
    const float* X  = (const float*)d_in[0];
    const float* E  = (const float*)d_in[1];
    const float* cc = (const float*)d_in[2];
    float* Out = (float*)d_out;
    float* ws  = (float*)d_ws;

    vq_prep_rows<<<dim3(NUM_EMB / 256), dim3(256), 0, stream>>>(E, ws);
    vq_prep_cols<<<dim3(1), dim3(1024), 0, stream>>>(E, ws);

    if (ws_size >= (size_t)WS_REQ_B) {
        __bf16* Ebf  = (__bf16*)((char*)d_ws + WS_EBF_B);
        __bf16* EbfT = (__bf16*)((char*)d_ws + WS_EBFT_B);
        vq_prep_bf16<<<dim3(NUM_EMB * EMB_DIM / 2048), dim3(256), 0, stream>>>(E, Ebf, EbfT);
        vq_main_async<<<dim3(NWG), dim3(128), 0, stream>>>(X, Ebf, EbfT, ws, Out);
    } else {
        vq_main_conv<<<dim3(NWG), dim3(128), 0, stream>>>(X, E, ws, Out);
    }

    vq_finalize<<<dim3(1), dim3(1), 0, stream>>>(ws, cc, Out);
}
